// TimeBlock_38139309588939
// MI455X (gfx1250) — matrix operand order repaired and checked by reading, not compiled, not run on hardware
//
#include <hip/hip_runtime.h>
#include <math.h>

typedef __attribute__((ext_vector_type(16))) __bf16 v16bf;
typedef __attribute__((ext_vector_type(8)))  float  v8f;

#define RTOT   65536   // B*L rows
#define DMODEL 512
#define NSRU   1536    // 3*H
#define FFDIM  2048
#define SEQL   4096
#define BATCH  16
#define NCHUNK 16
#define CHLEN  256     // SEQL / NCHUNK

static __device__ __forceinline__ __bf16 f2bf(float f) {
  union { float f; unsigned u; } a; a.f = f;
  unsigned u = a.u;
  unsigned r = u + 0x7FFFu + ((u >> 16) & 1u);   // round-to-nearest-even
  union { unsigned short s; __bf16 b; } o; o.s = (unsigned short)(r >> 16);
  return o.b;
}
static __device__ __forceinline__ float sigmoidf_(float v) {
  return 1.0f / (1.0f + __expf(-v));
}
static __device__ __forceinline__ float gelu_erf(float v) {
  return 0.5f * v * (1.0f + erff(v * 0.70710678118654752f));
}

// ---------------------------------------------------------------------------
// Pack fp32 weight (K x N, row major) into bf16 WMMA B-fragment layout:
// tile (ktile:32, ntile:16); within tile: lane = n%16 + 16*((kk/8)%2), elem = kk%8 + 8*(kk/16)
// Fragment for one (ntile,ktile) = 32 lanes * 16 bf16 = 1 KB, contiguous.
// ---------------------------------------------------------------------------
__global__ __launch_bounds__(256) void pack_b16(const float* __restrict__ W,
                                                __bf16* __restrict__ out, int K, int N) {
  int tid = blockIdx.x * 256 + threadIdx.x;
  if (tid >= K * N) return;
  int k = tid / N, n = tid % N;
  int kt = k >> 5, kk = k & 31;
  int nt = n >> 4, nl = n & 15;
  int lane = nl + (((kk >> 3) & 1) << 4);
  int elem = (kk & 7) + ((kk >> 4) << 3);
  size_t dst = (((size_t)nt * (size_t)(K >> 5) + (size_t)kt) << 9) + (size_t)(lane << 4) + (size_t)elem;
  out[dst] = f2bf(W[(size_t)k * (size_t)N + (size_t)n]);
}

// ---------------------------------------------------------------------------
// GEMM1: u = x @ W_sru  (M=65536, K=512, N=1536), bf16 WMMA, f32 accum.
// Epilogue: cols [0,512)=z identity, [512,1024)=sigmoid(+b_f), [1024,1536)=sigmoid(+b_r)
// Block = 4 waves; tile M=16, N=256 (wave: 4 ntiles of 16 cols).
// B fragments for a k-step are batch-preloaded so s_wait_loadcnt tapers
// instead of hitting 0 before every WMMA.
// ---------------------------------------------------------------------------
__global__ __launch_bounds__(128) void gemm_sru(const float* __restrict__ x,
                                                const __bf16* __restrict__ wp,
                                                const float* __restrict__ bfv,
                                                const float* __restrict__ brv,
                                                float* __restrict__ u) {
  __shared__ __align__(32) __bf16 As[32][16];          // packed A fragment [lane][elem]
  const int t = threadIdx.x;
  const int wave = t >> 5, lane = t & 31;
  const int row0 = blockIdx.x << 4;
  const int nt0  = (blockIdx.y << 4) + (wave << 2);    // 16 ntiles/block, 4 per wave
  v8f acc[4] = {};
  for (int kt = 0; kt < 16; ++kt) {
    __syncthreads();
    // stage A: 16 rows x 32 k, fp32 -> bf16, scattered into per-lane layout
    int idx = t << 2;
    int m = idx >> 5, kk0 = idx & 31;
    const float4 v4 = *(const float4*)(x + (size_t)(row0 + m) * DMODEL + (size_t)((kt << 5) + kk0));
    float vv[4] = {v4.x, v4.y, v4.z, v4.w};
#pragma unroll
    for (int i = 0; i < 4; ++i) {
      int kk = kk0 + i;
      int ln = m + (((kk >> 3) & 1) << 4);
      int el = (kk & 7) + ((kk >> 4) << 3);
      As[ln][el] = f2bf(vv[i]);
    }
    __syncthreads();
    v16bf a = *(const v16bf*)&As[lane][0];
    v16bf bfrag[4];
#pragma unroll
    for (int i = 0; i < 4; ++i)
      bfrag[i] = *(const v16bf*)(wp + ((((size_t)(nt0 + i)) * 16 + (size_t)kt) << 9) + (size_t)(lane << 4));
#pragma unroll
    for (int i = 0; i < 4; ++i)
      acc[i] = __builtin_amdgcn_wmma_f32_16x16x32_bf16(false, a, false, bfrag[i], (short)0, acc[i], false, false);
  }
  const int rbase = row0 + ((lane >> 4) << 3);         // C layout: row = j + 8*(lane/16)
#pragma unroll
  for (int i = 0; i < 4; ++i) {
    int n = ((nt0 + i) << 4) + (lane & 15);
#pragma unroll
    for (int j = 0; j < 8; ++j) {
      float v = acc[i][j];
      if (n >= DMODEL)
        v = (n < 2 * DMODEL) ? sigmoidf_(v + bfv[n - DMODEL]) : sigmoidf_(v + brv[n - 2 * DMODEL]);
      u[(size_t)(rbase + j) * NSRU + (size_t)n] = v;
    }
  }
}

// ---------------------------------------------------------------------------
// Chunked linear-recurrence scan: c_t = f*c + (1-f)*z is affine; chunk composes
// as c_out = P*c_in + Q with P = prod f, Q = local scan from 0.
// ---------------------------------------------------------------------------
__global__ __launch_bounds__(256) void scan_p1(const float* __restrict__ u,
                                               float* __restrict__ P, float* __restrict__ Q) {
  int tid = blockIdx.x * 256 + threadIdx.x;            // BATCH*DMODEL*NCHUNK threads
  int h = tid & (DMODEL - 1);
  int rest = tid >> 9;
  int chunk = rest & (NCHUNK - 1);
  int b = rest >> 4;
  float p = 1.0f, q = 0.0f;
  size_t base = ((size_t)b * SEQL + (size_t)chunk * CHLEN) * NSRU;
  for (int s = 0; s < CHLEN; ++s) {
    float f = u[base + (size_t)s * NSRU + DMODEL + h];
    float z = u[base + (size_t)s * NSRU + h];
    q = f * q + (1.0f - f) * z;
    p *= f;
  }
  int ch = b * DMODEL + h;
  P[(size_t)ch * NCHUNK + chunk] = p;
  Q[(size_t)ch * NCHUNK + chunk] = q;
}

__global__ __launch_bounds__(256) void scan_p2(const float* __restrict__ P,
                                               const float* __restrict__ Q,
                                               float* __restrict__ Cs) {
  int ch = blockIdx.x * 256 + threadIdx.x;             // BATCH*DMODEL channels
  float c = 0.0f;
  for (int k = 0; k < NCHUNK; ++k) {
    Cs[(size_t)ch * NCHUNK + k] = c;
    c = P[(size_t)ch * NCHUNK + k] * c + Q[(size_t)ch * NCHUNK + k];
  }
}

__global__ __launch_bounds__(256) void scan_p3(const float* __restrict__ u,
                                               const float* __restrict__ x,
                                               const float* __restrict__ Cs,
                                               float* __restrict__ xa) {
  int tid = blockIdx.x * 256 + threadIdx.x;
  int h = tid & (DMODEL - 1);
  int rest = tid >> 9;
  int chunk = rest & (NCHUNK - 1);
  int b = rest >> 4;
  float c = Cs[(size_t)(b * DMODEL + h) * NCHUNK + chunk];
  size_t rbase = (size_t)b * SEQL + (size_t)chunk * CHLEN;
  for (int s = 0; s < CHLEN; ++s) {
    size_t ru = (rbase + s) * NSRU;
    float f = u[ru + DMODEL + h];
    float z = u[ru + h];
    float r = u[ru + 2 * DMODEL + h];
    c = f * c + (1.0f - f) * z;
    float xv = x[(rbase + s) * DMODEL + h];
    xa[(rbase + s) * DMODEL + h] = xv + r * c + (1.0f - r) * xv;   // x + highway(sru)
  }
}

// ---------------------------------------------------------------------------
// LayerNorm over D=512: one wave32 per row, 16 values/lane, shfl_xor reduce.
// ---------------------------------------------------------------------------
__global__ __launch_bounds__(256) void layernorm_k(const float* __restrict__ in,
                                                   const float* __restrict__ g,
                                                   const float* __restrict__ bta,
                                                   float* __restrict__ out, int nrows) {
  int wave = threadIdx.x >> 5, lane = threadIdx.x & 31;
  int row = blockIdx.x * 8 + wave;
  if (row >= nrows) return;
  const float* p = in + (size_t)row * DMODEL + (size_t)(lane << 4);
  float vals[16];
#pragma unroll
  for (int i = 0; i < 4; ++i) {
    float4 v = ((const float4*)p)[i];
    vals[4 * i + 0] = v.x; vals[4 * i + 1] = v.y; vals[4 * i + 2] = v.z; vals[4 * i + 3] = v.w;
  }
  float s = 0.0f, s2 = 0.0f;
#pragma unroll
  for (int i = 0; i < 16; ++i) { s += vals[i]; s2 += vals[i] * vals[i]; }
#pragma unroll
  for (int m = 16; m >= 1; m >>= 1) {                  // wave32 butterfly
    s  += __shfl_xor(s,  m, 32);
    s2 += __shfl_xor(s2, m, 32);
  }
  float mean = s * (1.0f / DMODEL);
  float var  = s2 * (1.0f / DMODEL) - mean * mean;
  float inv  = rsqrtf(var + 1e-5f);
  float* q = out + (size_t)row * DMODEL + (size_t)(lane << 4);
#pragma unroll
  for (int i = 0; i < 16; ++i)
    q[i] = (vals[i] - mean) * inv * g[(lane << 4) + i] + bta[(lane << 4) + i];
}

// ---------------------------------------------------------------------------
// Fused FFN: x_c = x_b + gelu(x_b@W1 + b1)@W2 + b2 ; h never hits HBM.
// Block = 4 waves, M tile = 16 rows. x_b tile packed bf16 in LDS (16 KB).
// Stage 1 pipelines the W1 fragment stream (depth 2) and prefetches the next
// hidden chunk; stage 2 batches W2 fragment loads in groups of 4 so waits
// taper instead of serializing before each WMMA.
// ---------------------------------------------------------------------------
__global__ __launch_bounds__(128) void ffn_fused(const float* __restrict__ xb,
                                                 const __bf16* __restrict__ w1p,
                                                 const float* __restrict__ b1,
                                                 const __bf16* __restrict__ w2p,
                                                 const float* __restrict__ b2,
                                                 float* __restrict__ xc) {
  __shared__ __align__(32) __bf16 XA[16][32][16];      // x_b tile: 16 ktiles of A frags
  __shared__ __align__(32) __bf16 HA[2][32][16];       // h chunk: 2 ktiles of A frags
  const int t = threadIdx.x;
  const int wave = t >> 5, lane = t & 31;
  const int row0 = blockIdx.x << 4;

  // stage x_b tile (16 rows x 512) -> packed bf16 A fragments
#pragma unroll
  for (int i = 0; i < 16; ++i) {
    int idx4 = ((t << 4) + i) << 2;                    // 4 consecutive k, same row
    int m = idx4 >> 9, k0 = idx4 & 511;
    const float4 v4 = *(const float4*)(xb + (size_t)(row0 + m) * DMODEL + (size_t)k0);
    float vv[4] = {v4.x, v4.y, v4.z, v4.w};
#pragma unroll
    for (int i2 = 0; i2 < 4; ++i2) {
      int k = k0 + i2;
      int kt = k >> 5, kk = k & 31;
      int ln = m + (((kk >> 3) & 1) << 4);
      int el = (kk & 7) + ((kk >> 4) << 3);
      XA[kt][ln][el] = f2bf(vv[i2]);
    }
  }
  __syncthreads();

  v8f accO[8] = {};
  for (int hc = 0; hc < FFDIM / 64; ++hc) {
    // ---- stage 1: this wave's 16x16 h subtile, cols [hc*64 + wave*16, +16)
    v8f acc1 = {};
    const int nt1 = (hc << 2) + wave;
    const __bf16* w1f = w1p + (((size_t)nt1 * 16) << 9) + (size_t)(lane << 4);
    // prefetch next chunk's W1 fragment region into cache
    if (hc + 1 < FFDIM / 64)
      __builtin_prefetch(w1p + (((size_t)(nt1 + 4) * 16) << 9) + (size_t)(lane << 4), 0, 3);
    v16bf bcur = *(const v16bf*)w1f;
#pragma unroll
    for (int kt = 0; kt < 16; ++kt) {
      v16bf a = *(const v16bf*)&XA[kt][lane][0];
      v16bf bnext = bcur;
      if (kt < 15) bnext = *(const v16bf*)(w1f + ((size_t)(kt + 1) << 9));
      acc1 = __builtin_amdgcn_wmma_f32_16x16x32_bf16(false, a, false, bcur, (short)0, acc1, false, false);
      bcur = bnext;
    }
    // bias + gelu, repack into HA as GEMM2 A-fragments (h cols become K)
    int coll = (hc << 6) + (wave << 4) + (lane & 15);
    float bb = b1[coll];
    int kkb = ((wave << 4) + (lane & 15)) & 31;        // k within 32-wide ktile
    int ktl = wave >> 1;                               // which of the 2 local ktiles
    int m8  = (lane >> 4) << 3;
#pragma unroll
    for (int j = 0; j < 8; ++j) {
      float hv = gelu_erf(acc1[j] + bb);
      int m  = j + m8;
      int ln = m + (((kkb >> 3) & 1) << 4);
      int el = (kkb & 7) + ((kkb >> 4) << 3);
      HA[ktl][ln][el] = f2bf(hv);
    }
    __syncthreads();
    // ---- stage 2: accO += h_chunk @ W2[chunk rows, wave's 128 cols]
#pragma unroll
    for (int kt2 = 0; kt2 < 2; ++kt2) {
      v16bf a2 = *(const v16bf*)&HA[kt2][lane][0];
      int gkt = (hc << 1) + kt2;                       // global ktile in W2 (K=2048 -> 64)
#pragma unroll
      for (int cg = 0; cg < 2; ++cg) {                 // groups of 4 coltiles
        v16bf bf2[4];
#pragma unroll
        for (int ci = 0; ci < 4; ++ci) {
          int nt2 = (wave << 3) + (cg << 2) + ci;
          bf2[ci] = *(const v16bf*)(w2p + ((((size_t)nt2) * 64 + (size_t)gkt) << 9) + (size_t)(lane << 4));
        }
#pragma unroll
        for (int ci = 0; ci < 4; ++ci) {
          int ct = (cg << 2) + ci;
          accO[ct] = __builtin_amdgcn_wmma_f32_16x16x32_bf16(false, a2, false, bf2[ci], (short)0, accO[ct], false, false);
        }
      }
    }
    __syncthreads();
  }
  // epilogue: + b2 + residual x_b
#pragma unroll
  for (int ct = 0; ct < 8; ++ct) {
    int col = (wave << 7) + (ct << 4) + (lane & 15);
    float bb2 = b2[col];
#pragma unroll
    for (int j = 0; j < 8; ++j) {
      int row = row0 + j + ((lane >> 4) << 3);
      float v = accO[ct][j] + bb2 + xb[(size_t)row * DMODEL + (size_t)col];
      xc[(size_t)row * DMODEL + (size_t)col] = v;
    }
  }
}

// ---------------------------------------------------------------------------
extern "C" void kernel_launch(void* const* d_in, const int* in_sizes, int n_in,
                              void* d_out, int out_size, void* d_ws, size_t ws_size,
                              hipStream_t stream) {
  (void)in_sizes; (void)n_in; (void)out_size; (void)ws_size;
  const float* x    = (const float*)d_in[0];
  const float* Wsru = (const float*)d_in[1];
  const float* b_f  = (const float*)d_in[2];
  const float* b_r  = (const float*)d_in[3];
  const float* ln1g = (const float*)d_in[4];
  const float* ln1b = (const float*)d_in[5];
  const float* W1   = (const float*)d_in[6];
  const float* b1   = (const float*)d_in[7];
  const float* W2   = (const float*)d_in[8];
  const float* b2   = (const float*)d_in[9];
  const float* ln2g = (const float*)d_in[10];
  const float* ln2b = (const float*)d_in[11];
  float* out = (float*)d_out;

  char* base = (char*)d_ws;
  size_t off = 0;
  auto alloc = [&](size_t bytes) -> void* {
    void* p = base + off;
    off = (off + bytes + 255) & ~(size_t)255;
    return p;
  };
  __bf16* wsruP = (__bf16*)alloc((size_t)DMODEL * NSRU * 2);
  __bf16* w1P   = (__bf16*)alloc((size_t)DMODEL * FFDIM * 2);
  __bf16* w2P   = (__bf16*)alloc((size_t)FFDIM * DMODEL * 2);
  float*  Pb    = (float*)alloc((size_t)BATCH * DMODEL * NCHUNK * 4);
  float*  Qb    = (float*)alloc((size_t)BATCH * DMODEL * NCHUNK * 4);
  float*  Cs    = (float*)alloc((size_t)BATCH * DMODEL * NCHUNK * 4);
  float*  xa    = (float*)alloc((size_t)RTOT * DMODEL * 4);
  float*  xbuf  = (float*)alloc((size_t)RTOT * DMODEL * 4);
  float*  u     = (float*)alloc((size_t)RTOT * NSRU * 4);
  float*  xc    = u;   // alias: u is dead after scan_p3

  pack_b16<<<(DMODEL * NSRU + 255) / 256, 256, 0, stream>>>(Wsru, wsruP, DMODEL, NSRU);
  pack_b16<<<(DMODEL * FFDIM + 255) / 256, 256, 0, stream>>>(W1, w1P, DMODEL, FFDIM);
  pack_b16<<<(FFDIM * DMODEL + 255) / 256, 256, 0, stream>>>(W2, w2P, FFDIM, DMODEL);

  gemm_sru<<<dim3(RTOT / 16, NSRU / 256), 128, 0, stream>>>(x, wsruP, b_f, b_r, u);

  scan_p1<<<(BATCH * DMODEL * NCHUNK) / 256, 256, 0, stream>>>(u, Pb, Qb);
  scan_p2<<<(BATCH * DMODEL) / 256, 256, 0, stream>>>(Pb, Qb, Cs);
  scan_p3<<<(BATCH * DMODEL * NCHUNK) / 256, 256, 0, stream>>>(u, x, Cs, xa);

  layernorm_k<<<RTOT / 8, 256, 0, stream>>>(xa, ln1g, ln1b, xbuf, RTOT);

  ffn_fused<<<RTOT / 16, 128, 0, stream>>>(xbuf, w1P, b1, w2P, b2, xc);

  layernorm_k<<<RTOT / 8, 256, 0, stream>>>(xc, ln2g, ln2b, out, RTOT);
}
